// GlobalLocalTransformerLayers_58188216926783
// MI455X (gfx1250) — compile-verified, matrix operand order checked by reading, not run on hardware
//
#include <hip/hip_runtime.h>
#include <hip/hip_bf16.h>

// ---------------------------------------------------------------------------
// Types for CDNA5 WMMA (wave32, 16x16x32 bf16 -> f32)
// ---------------------------------------------------------------------------
typedef __attribute__((ext_vector_type(16))) __bf16 v16bf;
typedef __attribute__((ext_vector_type(8)))  __bf16 v8bf;
typedef __attribute__((ext_vector_type(4)))  __bf16 v4bf;
typedef __attribute__((ext_vector_type(8)))  float  v8f;
typedef __attribute__((ext_vector_type(4)))  float  v4f;

#define WMMA_BF16(a, b, c) \
  __builtin_amdgcn_wmma_f32_16x16x32_bf16(false, (a), false, (b), (short)0, (c), false, false)

static __device__ __forceinline__ v16bf cat8(v8bf lo, v8bf hi) {
  return __builtin_shufflevector(lo, hi, 0, 1, 2, 3, 4, 5, 6, 7, 8, 9, 10, 11, 12, 13, 14, 15);
}

// Build a bf16 A/B fragment from an fp32 LDS row (32 K-values / row).
// Per-lane layout for 16x16x32: lane<16 -> K {0..7, 16..23}, lane>=16 -> K {8..15, 24..31}.
static __device__ __forceinline__ v16bf cvt_frag(const float* __restrict__ p, int half) {
  v16bf r;
#pragma unroll
  for (int e = 0; e < 8; ++e) r[e] = (__bf16)p[half * 8 + e];
#pragma unroll
  for (int e = 0; e < 8; ++e) r[8 + e] = (__bf16)p[16 + half * 8 + e];
  return r;
}

// CDNA5 async copy: global -> LDS, 16 bytes per lane, tracked by ASYNCcnt.
static __device__ __forceinline__ void async_load_b128(const void* gaddr, void* lds) {
  unsigned l = (unsigned)(size_t)lds;               // generic LDS ptr low 32 bits = LDS offset
  unsigned long long g = (unsigned long long)(size_t)gaddr;
  asm volatile("global_load_async_to_lds_b128 %0, %1, off" :: "v"(l), "v"(g) : "memory");
}
static __device__ __forceinline__ void wait_async0() {
  asm volatile("s_wait_asynccnt 0x0" ::: "memory");
}
static __device__ __forceinline__ void wait_async4() {
  asm volatile("s_wait_asynccnt 0x4" ::: "memory");
}
static __device__ __forceinline__ void wait_async8() {
  asm volatile("s_wait_asynccnt 0x8" ::: "memory");
}

// ---------------------------------------------------------------------------
// Problem dimensions (fixed by the reference)
// ---------------------------------------------------------------------------
constexpr int cB   = 2;
constexpr int cL   = 4096;
constexpr int cG   = 256;
constexpr int cD   = 768;
constexpr int cH   = 12;
constexpr int cNL  = 2;
constexpr int cR   = 128;
constexpr int cDH  = 64;
constexpr int cDFF = 3072;
constexpr int cV   = 257;     // 2R+1
constexpr int cNB  = 32;      // L / R
constexpr int cGL  = 4352;    // G + L
constexpr float cSCALE = 0.125f;  // 1/sqrt(64)

// GEMM flags
constexpr int F_TRB  = 1;  // B stored as [N][K] (template selector)
constexpr int F_GELU = 2;  // tanh-approx GELU epilogue
constexpr int F_ACC  = 4;  // C += result

// ---------------------------------------------------------------------------
// Generic batched WMMA GEMM:  C = act(alpha * A @ B (+bias) (+C))
// A: [M][K] fp32, row stride lda; REQUIREMENT: M % 128 == 0 (no M guard).
// B: [K][N] (or [N][K] if TRB) fp32, row stride ldb.
// N is guarded ONLY at the store: out-of-range B rows/cols are fetched with
// clamped (always-valid) addresses — the garbage lands exclusively in output
// columns >= N which are never written. This keeps staging branch-free and
// the async issue count uniform (8 loads/thread-step for TRB, 4 otherwise),
// enabling exact-count ASYNCcnt pipelining.
// K must be a multiple of 32; row strides multiples of 4 floats; N >= 4.
// Tile: 128(M) x 128(N) x 32(K), 256 threads = 8 waves (4M x 2N of 32x64).
// Double-buffered LDS; tile i+1 is async-copied while tile i computes;
// s_wait_asynccnt <= (count just issued) guarantees tile i has landed.
// ---------------------------------------------------------------------------
template <bool TRB>
__global__ __launch_bounds__(256) void wmma_gemm_t(
    const float* __restrict__ A, long long lda, long long sAo, long long sAi,
    const float* __restrict__ Bm, long long ldb, long long sBo, long long sBi,
    float* __restrict__ C, long long ldc, long long sCo, long long sCi,
    const float* __restrict__ bias, int M, int N, int K, int bInner,
    float alpha, int flags) {
  __shared__ __align__(16) float Asf[2][128 * 32];   // 2 x 16 KB, [m][k]
  __shared__ __align__(16) float Bsf[2][128 * 32];   // 2 x 16 KB, [n][k]

  int z  = blockIdx.z;
  int zo = z / bInner, zi = z % bInner;
  const float* Ab = A  + zo * sAo + zi * sAi;
  const float* Bp = Bm + zo * sBo + zi * sBi;
  float*       Cb = C  + zo * sCo + zi * sCi;

  int m0 = blockIdx.y * 128, n0 = blockIdx.x * 128;
  int t = threadIdx.x, lane = t & 31, w = t >> 5;
  int wM = (w & 3) * 32, wN = (w >> 2) * 64;
  int half = (lane >> 4) & 1, l15 = lane & 15;

  v8f vzero = {0.f, 0.f, 0.f, 0.f, 0.f, 0.f, 0.f, 0.f};
  v8f acc[2][4];
  for (int i = 0; i < 2; ++i)
    for (int j = 0; j < 4; ++j) acc[i][j] = vzero;

  auto stage = [&](int k0, int buf) {
    // A tile: 128x32 fp32, unconditional (M % 128 == 0), 4 async b128 / thread
#pragma unroll
    for (int i = 0; i < 4; ++i) {
      int idx4 = t + i * 256;             // 1024 float4s
      int r = idx4 >> 3;                  // 8 float4 per 32-float row
      int c4 = (idx4 & 7) * 4;
      async_load_b128(&Ab[(long long)(m0 + r) * lda + (k0 + c4)],
                      &Asf[buf][r * 32 + c4]);
    }
    if constexpr (TRB) {
      // B already [N][K]: clamp row (cols >= N never stored), 4 async b128 / thread
#pragma unroll
      for (int i = 0; i < 4; ++i) {
        int idx4 = t + i * 256;
        int r = idx4 >> 3;
        int c4 = (idx4 & 7) * 4;
        int nidx = n0 + r;
        if (nidx >= N) nidx = N - 1;      // clamped, always valid
        async_load_b128(&Bp[(long long)nidx * ldb + (k0 + c4)],
                        &Bsf[buf][r * 32 + c4]);
      }
    } else {
      // B [K][N]: vector-load along N (clamped aligned base), transpose-scatter.
#pragma unroll
      for (int i = 0; i < 4; ++i) {
        int idx4 = t + i * 256;
        int kk = idx4 >> 5;               // 32 float4 per 128-float N-row
        int n4 = (idx4 & 31) * 4;
        int nn = n0 + n4;
        int nncl = (nn + 3 < N) ? nn : ((N - 4) & ~3);  // aligned, always valid
        v4f v = *(const v4f*)&Bp[(long long)(k0 + kk) * ldb + nncl];
#pragma unroll
        for (int e = 0; e < 4; ++e) Bsf[buf][(n4 + e) * 32 + kk] = v[e];
      }
    }
  };

  const int nt = K >> 5;
  stage(0, 0);
  for (int it = 0; it < nt; ++it) {
    int cur = it & 1;
    bool more = (it + 1 < nt);
    if (more) stage((it + 1) << 5, cur ^ 1);   // prefetch next tile (other buffer)
    if (more) {
      if constexpr (TRB) wait_async8(); else wait_async4();
    } else {
      wait_async0();
    }
    __syncthreads();

    v16bf af[2], bfr[4];
#pragma unroll
    for (int i = 0; i < 2; ++i)
      af[i] = cvt_frag(&Asf[cur][(wM + i * 16 + l15) * 32], half);
#pragma unroll
    for (int j = 0; j < 4; ++j)
      bfr[j] = cvt_frag(&Bsf[cur][(wN + j * 16 + l15) * 32], half);
#pragma unroll
    for (int i = 0; i < 2; ++i)
#pragma unroll
      for (int j = 0; j < 4; ++j)
        acc[i][j] = WMMA_BF16(af[i], bfr[j], acc[i][j]);
    __syncthreads();
  }

  // ---- epilogue (the only place N is guarded) ----
#pragma unroll
  for (int i = 0; i < 2; ++i)
#pragma unroll
    for (int j = 0; j < 4; ++j)
#pragma unroll
      for (int e = 0; e < 8; ++e) {
        int m = m0 + wM + i * 16 + e + ((lane >= 16) ? 8 : 0);
        int n = n0 + wN + j * 16 + l15;
        if (n < N) {   // m always < M (M % 128 == 0)
          float v = acc[i][j][e] * alpha;
          if (bias) v += bias[n];
          long long off = (long long)m * ldc + n;
          if (flags & F_ACC) v += Cb[off];
          if (flags & F_GELU) {
            float u = v;
            v = 0.5f * u * (1.0f + tanhf(0.7978845608f * (u + 0.044715f * u * u * u)));
          }
          Cb[off] = v;
        }
      }
}

// ---------------------------------------------------------------------------
// Fused local (long-to-long + long-to-global) attention.
// One workgroup = (b, block n, head h, 32-query subtile qt). 128 thr = 4 waves.
// Keys: 5 tiles of 128 = 384 local (3 blocks) + 256 global. Joint softmax.
// QE[b][q][h][v] = Q . rel[v]  (precomputed GEMM) provides the rel-pos bias.
// ---------------------------------------------------------------------------
__global__ __launch_bounds__(128) void local_attn(
    const float* __restrict__ Q, const float* __restrict__ Kl,
    const float* __restrict__ Vl, const float* __restrict__ Kg,
    const float* __restrict__ Vg, const float* __restrict__ QE,
    float* __restrict__ ctx) {
  __shared__ __align__(16) __bf16 Qs[32 * 64];    // 4 KB
  __shared__ __align__(16) __bf16 Ks[128 * 64];   // 16 KB (K stage; V^T stage in phase 3)
  __shared__ __align__(16) __bf16 Sb[32 * 640];   // 40 KB score/prob buffer

  int x  = blockIdx.x;
  int qt = x & 3;
  int h  = (x >> 2) % cH;
  int n  = (x / (4 * cH)) % cNB;
  int b  = x / (4 * cH * cNB);

  int t = threadIdx.x, lane = t & 31, w = t >> 5;
  int half = (lane >> 4) & 1, l15 = lane & 15;

  v4f z4 = {0.f, 0.f, 0.f, 0.f};

  // stage Q subtile (32 x 64), unconditional b128 loads
#pragma unroll
  for (int i = 0; i < 4; ++i) {
    int idx4 = t + i * 128;               // 512 float4s
    int r = idx4 >> 4, c4 = (idx4 & 15) * 4;
    int q = n * cR + qt * 32 + r;
    v4f v = *(const v4f*)&Q[((long long)(b * cL + q) * cH + h) * cDH + c4];
    v4bf h4;
#pragma unroll
    for (int e = 0; e < 4; ++e) h4[e] = (__bf16)v[e];
    *(v4bf*)&Qs[r * 64 + c4] = h4;
  }

  v8f vzero = {0.f, 0.f, 0.f, 0.f, 0.f, 0.f, 0.f, 0.f};
  v8f accO[2];
  accO[0] = vzero;
  accO[1] = vzero;

  // ---------------- phase 1: scores (WMMA Q x K^T) ----------------
  for (int kt = 0; kt < 5; ++kt) {
    __syncthreads();  // protect Ks reuse (and Qs staging on first iter)
    {
      // Gather 128 keys x 64 dims: clamped-address b128 loads (no divergent waits),
      // buffered in registers, then converted + stored as b64.
      v4f vals[16];
#pragma unroll
      for (int i = 0; i < 16; ++i) {
        int idx4 = t + i * 128;           // 2048 float4s
        int r = idx4 >> 4, c4 = (idx4 & 15) * 4;
        if (kt < 3) {
          int kpos = n * cR - cR + kt * cR + r;
          bool valid = (kpos >= 0) && (kpos < cL);
          int kcl = kpos < 0 ? 0 : (kpos >= cL ? cL - 1 : kpos);
          v4f v = *(const v4f*)&Kl[((long long)(b * cL + kcl) * cH + h) * cDH + c4];
          vals[i] = valid ? v : z4;
        } else {
          int g = (kt - 3) * 128 + r;
          vals[i] = *(const v4f*)&Kg[((long long)(b * cG + g) * cH + h) * cDH + c4];
        }
      }
#pragma unroll
      for (int i = 0; i < 16; ++i) {
        int idx4 = t + i * 128;
        int r = idx4 >> 4, c4 = (idx4 & 15) * 4;
        v4bf h4;
#pragma unroll
        for (int e = 0; e < 4; ++e) h4[e] = (__bf16)vals[i][e];
        *(v4bf*)&Ks[r * 64 + c4] = h4;
      }
    }
    __syncthreads();

    v8f accS[2][2];
    accS[0][0] = vzero; accS[0][1] = vzero;
    accS[1][0] = vzero; accS[1][1] = vzero;
#pragma unroll
    for (int ks = 0; ks < 2; ++ks) {
      v16bf af[2], bfr[2];
#pragma unroll
      for (int i = 0; i < 2; ++i) {
        const __bf16* p = &Qs[(i * 16 + l15) * 64 + ks * 32];
        af[i] = cat8(*(const v8bf*)(p + half * 8), *(const v8bf*)(p + 16 + half * 8));
      }
#pragma unroll
      for (int j = 0; j < 2; ++j) {
        const __bf16* p = &Ks[(w * 32 + j * 16 + l15) * 64 + ks * 32];
        bfr[j] = cat8(*(const v8bf*)(p + half * 8), *(const v8bf*)(p + 16 + half * 8));
      }
#pragma unroll
      for (int i = 0; i < 2; ++i)
#pragma unroll
        for (int j = 0; j < 2; ++j)
          accS[i][j] = WMMA_BF16(af[i], bfr[j], accS[i][j]);
    }

    // bias + scale + mask, write to Sb.
    // The QE gather uses a clamped (always-valid) index AND the loaded value is
    // consumed on BOTH sides of the mask select (fmaf(bias, 0, -1e9) == -1e9 for
    // finite bias): this prevents the compiler from sinking the load into a
    // divergent branch, so the 32 independent gathers issue back-to-back.
#pragma unroll
    for (int i = 0; i < 2; ++i)
#pragma unroll
      for (int j = 0; j < 2; ++j)
#pragma unroll
        for (int e = 0; e < 8; ++e) {
          int m  = i * 16 + e + ((lane >= 16) ? 8 : 0);      // 0..31
          int kk = kt * 128 + w * 32 + j * 16 + l15;         // 0..639
          float s;
          if (kk < 3 * cR) {   // local keys: rel-pos bias + mask
            int qq     = qt * 32 + m;                        // 0..127 in block
            int relpos = kk - cR - qq;
            int kpos   = n * cR + kk - cR;
            int rid    = relpos + cR;
            rid = rid < 0 ? 0 : (rid > cV - 1 ? cV - 1 : rid);
            float bias = QE[((long long)(b * cL + n * cR + qq) * cH + h) * cV + rid];
            bool valid = (relpos >= -cR) && (relpos <= cR) && (kpos >= 0) && (kpos < cL);
            float masked = fmaf(bias, 0.0f, -1e9f);          // keeps 'bias' live unconditionally
            s = valid ? (accS[i][j][e] + bias) * cSCALE : masked;
          } else {             // global keys
            s = accS[i][j][e] * cSCALE;
          }
          Sb[m * 640 + kk] = (__bf16)s;
        }
  }
  __syncthreads();

  // ---------------- phase 2: joint softmax over 640 keys ----------------
  if (t < 32) {
    float mx = -1e30f;
    for (int k = 0; k < 640; ++k) {
      float v = (float)Sb[t * 640 + k];
      mx = fmaxf(mx, v);
    }
    float sum = 0.f;
    for (int k = 0; k < 640; ++k) {
      float e = __expf((float)Sb[t * 640 + k] - mx);
      sum += e;
      Sb[t * 640 + k] = (__bf16)e;
    }
    float inv = 1.f / sum;
    for (int k = 0; k < 640; ++k)
      Sb[t * 640 + k] = (__bf16)((float)Sb[t * 640 + k] * inv);
  }

  // ---------------- phase 3: ctx = P @ V (WMMA) ----------------
  for (int kt = 0; kt < 5; ++kt) {
    __syncthreads();
    {
      v4f vals[16];
#pragma unroll
      for (int i = 0; i < 16; ++i) {
        int idx4 = t + i * 128;
        int r = idx4 >> 4, c4 = (idx4 & 15) * 4;
        if (kt < 3) {
          int kpos = n * cR - cR + kt * cR + r;
          bool valid = (kpos >= 0) && (kpos < cL);
          int kcl = kpos < 0 ? 0 : (kpos >= cL ? cL - 1 : kpos);
          v4f v = *(const v4f*)&Vl[((long long)(b * cL + kcl) * cH + h) * cDH + c4];
          vals[i] = valid ? v : z4;
        } else {
          int g = (kt - 3) * 128 + r;
          vals[i] = *(const v4f*)&Vg[((long long)(b * cG + g) * cH + h) * cDH + c4];
        }
      }
      // transposed store: Ks holds V^T [d][key]
#pragma unroll
      for (int i = 0; i < 16; ++i) {
        int idx4 = t + i * 128;
        int r = idx4 >> 4, c4 = (idx4 & 15) * 4;
#pragma unroll
        for (int e = 0; e < 4; ++e) Ks[(c4 + e) * 128 + r] = (__bf16)vals[i][e];
      }
    }
    __syncthreads();

#pragma unroll
    for (int ks = 0; ks < 4; ++ks) {
      const __bf16* pb = &Ks[(w * 16 + l15) * 128 + ks * 32];
      v16bf bfr = cat8(*(const v8bf*)(pb + half * 8), *(const v8bf*)(pb + 16 + half * 8));
#pragma unroll
      for (int i = 0; i < 2; ++i) {
        const __bf16* pa = &Sb[(i * 16 + l15) * 640 + kt * 128 + ks * 32];
        v16bf af = cat8(*(const v8bf*)(pa + half * 8), *(const v8bf*)(pa + 16 + half * 8));
        accO[i] = WMMA_BF16(af, bfr, accO[i]);
      }
    }
  }

  // write ctx [B,L,H,DH]
#pragma unroll
  for (int i = 0; i < 2; ++i)
#pragma unroll
    for (int e = 0; e < 8; ++e) {
      int m = i * 16 + e + ((lane >= 16) ? 8 : 0);
      int q = n * cR + qt * 32 + m;
      int d = w * 16 + l15;
      ctx[((long long)(b * cL + q) * cH + h) * cDH + d] = accO[i][e];
    }
}

// ---------------------------------------------------------------------------
// Row softmax (with pre-scale) for the global-attention score matrix.
// ---------------------------------------------------------------------------
__global__ __launch_bounds__(256) void softmax_rows(float* __restrict__ S, int cols,
                                                    float scale) {
  __shared__ float red[256];
  long long base = (long long)blockIdx.x * cols;
  int t = threadIdx.x;

  float mx = -1e30f;
  for (int c = t; c < cols; c += 256) mx = fmaxf(mx, S[base + c] * scale);
  red[t] = mx;
  __syncthreads();
  for (int o = 128; o > 0; o >>= 1) {
    if (t < o) red[t] = fmaxf(red[t], red[t + o]);
    __syncthreads();
  }
  mx = red[0];
  __syncthreads();

  float sum = 0.f;
  for (int c = t; c < cols; c += 256) {
    float e = __expf(S[base + c] * scale - mx);
    S[base + c] = e;
    sum += e;
  }
  red[t] = sum;
  __syncthreads();
  for (int o = 128; o > 0; o >>= 1) {
    if (t < o) red[t] += red[t + o];
    __syncthreads();
  }
  float inv = 1.f / red[0];
  for (int c = t; c < cols; c += 256) S[base + c] *= inv;
}

// ---------------------------------------------------------------------------
// x = LayerNorm(x + a) * gamma + beta   (D = 768, one row per block)
// ---------------------------------------------------------------------------
__global__ __launch_bounds__(256) void add_ln(float* __restrict__ x,
                                              const float* __restrict__ a,
                                              const float* __restrict__ gamma,
                                              const float* __restrict__ beta) {
  __shared__ float red[256];
  long long base = (long long)blockIdx.x * cD;
  int t = threadIdx.x;

  float s0 = x[base + t]       + a[base + t];
  float s1 = x[base + t + 256] + a[base + t + 256];
  float s2 = x[base + t + 512] + a[base + t + 512];

  red[t] = s0 + s1 + s2;
  __syncthreads();
  for (int o = 128; o > 0; o >>= 1) {
    if (t < o) red[t] += red[t + o];
    __syncthreads();
  }
  float mean = red[0] * (1.0f / cD);
  __syncthreads();

  float d0 = s0 - mean, d1 = s1 - mean, d2 = s2 - mean;
  red[t] = d0 * d0 + d1 * d1 + d2 * d2;
  __syncthreads();
  for (int o = 128; o > 0; o >>= 1) {
    if (t < o) red[t] += red[t + o];
    __syncthreads();
  }
  float inv = rsqrtf(red[0] * (1.0f / cD) + 1e-12f);

  x[base + t]       = d0 * inv * gamma[t]       + beta[t];
  x[base + t + 256] = d1 * inv * gamma[t + 256] + beta[t + 256];
  x[base + t + 512] = d2 * inv * gamma[t + 512] + beta[t + 512];
}

// ---------------------------------------------------------------------------
// out[B, L+G, D] = concat(lx, gx, axis=1)
// ---------------------------------------------------------------------------
__global__ void final_concat(const float* __restrict__ lx, const float* __restrict__ gx,
                             float* __restrict__ out, long long total) {
  long long i = (long long)blockIdx.x * 256 + threadIdx.x;
  if (i >= total) return;
  int c = (int)(i % cD);
  long long r = i / cD;
  int row = (int)(r % (cL + cG));
  int b   = (int)(r / (cL + cG));
  float v = (row < cL) ? lx[((long long)b * cL + row) * cD + c]
                       : gx[((long long)b * cG + (row - cL)) * cD + c];
  out[i] = v;
}

// ---------------------------------------------------------------------------
// Host orchestration
// ---------------------------------------------------------------------------
static inline void launch_gemm(hipStream_t s,
    const float* A, long long lda, long long sAo, long long sAi,
    const float* Bm, long long ldb, long long sBo, long long sBi,
    float* C, long long ldc, long long sCo, long long sCi,
    const float* bias, int M, int N, int K, int bOuter, int bInner,
    float alpha, int flags) {
  dim3 grid((N + 127) / 128, (M + 127) / 128, bOuter * bInner);
  if (flags & F_TRB) {
    wmma_gemm_t<true><<<grid, dim3(256), 0, s>>>(
        A, lda, sAo, sAi, Bm, ldb, sBo, sBi, C, ldc, sCo, sCi,
        bias, M, N, K, bInner, alpha, flags);
  } else {
    wmma_gemm_t<false><<<grid, dim3(256), 0, s>>>(
        A, lda, sAo, sAi, Bm, ldb, sBo, sBi, C, ldc, sCo, sCi,
        bias, M, N, K, bInner, alpha, flags);
  }
}

extern "C" void kernel_launch(void* const* d_in, const int* in_sizes, int n_in,
                              void* d_out, int out_size, void* d_ws, size_t ws_size,
                              hipStream_t stream) {
  (void)in_sizes; (void)n_in; (void)out_size; (void)ws_size;

  const float* in_long = (const float*)d_in[0];
  const float* in_glob = (const float*)d_in[1];
  const float* Wq   = (const float*)d_in[2];
  const float* Wk   = (const float*)d_in[3];
  const float* Wv   = (const float*)d_in[4];
  const float* bqkv = (const float*)d_in[5];
  const float* Wol  = (const float*)d_in[6];
  const float* Wog  = (const float*)d_in[7];
  const float* bout = (const float*)d_in[8];
  const float* rel  = (const float*)d_in[9];
  const float* W1   = (const float*)d_in[10];
  const float* b1   = (const float*)d_in[11];
  const float* W2   = (const float*)d_in[12];
  const float* b2   = (const float*)d_in[13];
  const float* lng  = (const float*)d_in[14];
  const float* lnb  = (const float*)d_in[15];

  float* ws = (float*)d_ws;
  size_t off = 0;
  auto alloc = [&](size_t nElem) { float* p = ws + off; off += nElem; return p; };

  const size_t ML = (size_t)cB * cL * cD;  // 6.29M floats
  const size_t MG = (size_t)cB * cG * cD;  // 0.39M floats

  float* lx   = alloc(ML);
  float* gx   = alloc(MG);
  float* Ql   = alloc(ML);
  float* Kl   = alloc(ML);
  float* Vl   = alloc(ML);
  float* Qg   = alloc(MG);
  float* Kg   = alloc(MG);
  float* Vg   = alloc(MG);
  float* ctxL = alloc(ML);   // also reused as FFN output
  float* laB  = alloc(ML);
  float* gctx = alloc(MG);
  float* gaB  = alloc(MG);
  // shared scratch: max(QE = B*L*H*V, Sg = B*H*G*GL, ffnT = B*L*DFF)
  const size_t scrN = (size_t)cB * cH * cG * cGL;  // 26.7M floats (largest)
  float* SCR = alloc(scrN);

  hipMemcpyAsync(lx, in_long, ML * sizeof(float), hipMemcpyDeviceToDevice, stream);
  hipMemcpyAsync(gx, in_glob, MG * sizeof(float), hipMemcpyDeviceToDevice, stream);

  for (int i = 0; i < cNL; ++i) {
    const float* wq   = Wq  + (size_t)i * cD * cD;
    const float* wk   = Wk  + (size_t)i * cD * cD;
    const float* wv   = Wv  + (size_t)i * cD * cD;
    const float* bq   = bqkv + (size_t)i * 3 * cD;
    const float* wol  = Wol + (size_t)i * cD * cD;
    const float* wog  = Wog + (size_t)i * cD * cD;
    const float* bo   = bout + (size_t)i * 2 * cD;
    const float* reli = rel + (size_t)i * cV * cH * cDH;
    const float* w1   = W1 + (size_t)i * cD * cDFF;
    const float* b1i  = b1 + (size_t)i * cDFF;
    const float* w2   = W2 + (size_t)i * cDFF * cD;
    const float* b2i  = b2 + (size_t)i * cD;

    // ---- QKV projections (shared weights for long + global streams) ----
    launch_gemm(stream, lx, cD, 0, 0, wq, cD, 0, 0, Ql, cD, 0, 0,
                bq + 0 * cD, cB * cL, cD, cD, 1, 1, 1.f, 0);
    launch_gemm(stream, lx, cD, 0, 0, wk, cD, 0, 0, Kl, cD, 0, 0,
                bq + 1 * cD, cB * cL, cD, cD, 1, 1, 1.f, 0);
    launch_gemm(stream, lx, cD, 0, 0, wv, cD, 0, 0, Vl, cD, 0, 0,
                bq + 2 * cD, cB * cL, cD, cD, 1, 1, 1.f, 0);
    launch_gemm(stream, gx, cD, 0, 0, wq, cD, 0, 0, Qg, cD, 0, 0,
                bq + 0 * cD, cB * cG, cD, cD, 1, 1, 1.f, 0);
    launch_gemm(stream, gx, cD, 0, 0, wk, cD, 0, 0, Kg, cD, 0, 0,
                bq + 1 * cD, cB * cG, cD, cD, 1, 1, 1.f, 0);
    launch_gemm(stream, gx, cD, 0, 0, wv, cD, 0, 0, Vg, cD, 0, 0,
                bq + 2 * cD, cB * cG, cD, cD, 1, 1, 1.f, 0);

    // ---- QE = Q . rel^T  (batched over (b, h); rel shared across b) ----
    launch_gemm(stream,
                Ql, cD, (long long)cL * cD, cDH,
                reli, cD, 0, cDH,
                SCR, (long long)cH * cV, (long long)cL * cH * cV, cV,
                nullptr, cL, cV, cDH, cB, cH, 1.f, F_TRB);

    // ---- fused local attention ----
    local_attn<<<dim3(cB * cNB * cH * 4), dim3(128), 0, stream>>>(
        Ql, Kl, Vl, Kg, Vg, SCR, ctxL);

    // ---- long output projection ----
    launch_gemm(stream, ctxL, cD, 0, 0, wol, cD, 0, 0, laB, cD, 0, 0,
                bo + 0 * cD, cB * cL, cD, cD, 1, 1, 1.f, 0);

    // ---- global attention: scores, softmax, context ----
    float* Sg = SCR;  // QE no longer needed
    launch_gemm(stream,
                Qg, cD, (long long)cG * cD, cDH,
                Kg, cD, (long long)cG * cD, cDH,
                Sg, cGL, (long long)cH * cG * cGL, (long long)cG * cGL,
                nullptr, cG, cG, cDH, cB, cH, 1.f, F_TRB);
    launch_gemm(stream,
                Qg, cD, (long long)cG * cD, cDH,
                Kl, cD, (long long)cL * cD, cDH,
                Sg + cG, cGL, (long long)cH * cG * cGL, (long long)cG * cGL,
                nullptr, cG, cL, cDH, cB, cH, 1.f, F_TRB);
    softmax_rows<<<dim3(cB * cH * cG), dim3(256), 0, stream>>>(Sg, cGL, cSCALE);
    launch_gemm(stream,
                Sg, cGL, (long long)cH * cG * cGL, (long long)cG * cGL,
                Vg, cD, (long long)cG * cD, cDH,
                gctx, cD, (long long)cG * cD, cDH,
                nullptr, cG, cDH, cG, cB, cH, 1.f, 0);
    launch_gemm(stream,
                Sg + cG, cGL, (long long)cH * cG * cGL, (long long)cG * cGL,
                Vl, cD, (long long)cL * cD, cDH,
                gctx, cD, (long long)cG * cD, cDH,
                nullptr, cG, cDH, cL, cB, cH, 1.f, F_ACC);
    launch_gemm(stream, gctx, cD, 0, 0, wog, cD, 0, 0, gaB, cD, 0, 0,
                bo + 1 * cD, cB * cG, cD, cD, 1, 1, 1.f, 0);

    // ---- residual + LN ----
    add_ln<<<dim3(cB * cL), dim3(256), 0, stream>>>(
        lx, laB, lng + (size_t)(i * 4 + 0) * cD, lnb + (size_t)(i * 4 + 0) * cD);
    add_ln<<<dim3(cB * cG), dim3(256), 0, stream>>>(
        gx, gaB, lng + (size_t)(i * 4 + 1) * cD, lnb + (size_t)(i * 4 + 1) * cD);

    // ---- FFN (long) ----
    launch_gemm(stream, lx, cD, 0, 0, w1, cDFF, 0, 0, SCR, cDFF, 0, 0,
                b1i, cB * cL, cDFF, cD, 1, 1, 1.f, F_GELU);
    launch_gemm(stream, SCR, cDFF, 0, 0, w2, cD, 0, 0, ctxL, cD, 0, 0,
                b2i, cB * cL, cD, cDFF, 1, 1, 1.f, 0);
    add_ln<<<dim3(cB * cL), dim3(256), 0, stream>>>(
        lx, ctxL, lng + (size_t)(i * 4 + 2) * cD, lnb + (size_t)(i * 4 + 2) * cD);

    // ---- FFN (global) ----
    launch_gemm(stream, gx, cD, 0, 0, w1, cDFF, 0, 0, SCR, cDFF, 0, 0,
                b1i, cB * cG, cDFF, cD, 1, 1, 1.f, F_GELU);
    launch_gemm(stream, SCR, cDFF, 0, 0, w2, cD, 0, 0, ctxL, cD, 0, 0,
                b2i, cB * cG, cD, cDFF, 1, 1, 1.f, 0);
    add_ln<<<dim3(cB * cG), dim3(256), 0, stream>>>(
        gx, ctxL, lng + (size_t)(i * 4 + 3) * cD, lnb + (size_t)(i * 4 + 3) * cD);
  }

  // ---- output: concat([lx, gx], axis=1) ----
  long long total = (long long)cB * (cL + cG) * cD;
  final_concat<<<dim3((unsigned)((total + 255) / 256)), dim3(256), 0, stream>>>(
      lx, gx, (float*)d_out, total);
}
